// Linear_60404420051615
// MI455X (gfx1250) — matrix operand order repaired and checked by reading, not compiled, not run on hardware
//
#include <hip/hip_runtime.h>
#include <hip/hip_bf16.h>

typedef __attribute__((ext_vector_type(16))) __bf16 v16bf;
typedef __attribute__((ext_vector_type(8)))  __bf16 v8bf;
typedef __attribute__((ext_vector_type(4)))  __bf16 v4bf;
typedef __attribute__((ext_vector_type(8)))  float  v8f;

#define D_IN   1024
#define D_OUT  1024
#define KTILE  32
#define MTILE  128
#define NTILE  128
#define LDSS   40      // LDS row stride in bf16 elements (80B) -> conflict-free b128 reads
#define KSTEPS (D_IN / KTILE)

// ---------------------------------------------------------------------------
// Kernel 1: fold all LoRA adapters into the base weight, emit bf16 W_eff.
// W_eff[n,k] = W_base[n,k] + sum_i lambda_i*scale_i * sum_j B_i[n,j]*A_i[j,k]
// ---------------------------------------------------------------------------
__global__ __launch_bounds__(256) void fold_weights(
    const float* __restrict__ Wb,
    const float* __restrict__ A32, const float* __restrict__ B32,
    const float* __restrict__ A16, const float* __restrict__ B16,
    const float* __restrict__ A8,  const float* __restrict__ B8,
    const float* __restrict__ A4,  const float* __restrict__ B4,
    const float* __restrict__ A2,  const float* __restrict__ B2,
    const float* __restrict__ A1,  const float* __restrict__ B1,
    const float* __restrict__ lambdas, const float* __restrict__ scaling,
    unsigned short* __restrict__ Wq)
{
    const int idx = blockIdx.x * 256 + threadIdx.x;   // k contiguous within block
    const int n = idx >> 10;
    const int k = idx & 1023;

    float acc = Wb[idx];

    const float* As[6] = {A32, A16, A8, A4, A2, A1};
    const float* Bs[6] = {B32, B16, B8, B4, B2, B1};
    const int    rs[6] = {32, 16, 8, 4, 2, 1};

    #pragma unroll
    for (int i = 0; i < 6; ++i) {
        const float s = lambdas[i] * scaling[i];
        const float* __restrict__ A = As[i];
        const float* __restrict__ B = Bs[i];
        const int r = rs[i];
        float sum = 0.f;
        for (int j = 0; j < r; ++j)
            sum += B[n * r + j] * A[j * D_IN + k];   // B: SGPR broadcast, A: coalesced
        acc += s * sum;
    }
    __bf16 h = (__bf16)acc;
    Wq[idx] = __builtin_bit_cast(unsigned short, h);
}

// ---------------------------------------------------------------------------
// Async global->LDS copy of one 16B chunk per lane (CDNA5, ASYNCcnt-tracked).
// lds_off: byte offset within workgroup LDS (low 32 bits of generic pointer,
// per ISA aperture rule).
// ---------------------------------------------------------------------------
__device__ __forceinline__ void async_copy_b128(unsigned lds_off, const void* gaddr) {
    asm volatile("global_load_async_to_lds_b128 %0, %1, off"
                 :: "v"(lds_off), "v"(gaddr)
                 : "memory");
}
__device__ __forceinline__ void wait_async0() {
    asm volatile("s_wait_asynccnt 0x0" ::: "memory");
}

// ---------------------------------------------------------------------------
// Kernel 2: Y[M,N] = Xbf16[M,K] * W_eff_bf16[N,K]^T + bias, f32 accumulate.
// 128x128 block tile, 8 waves, each wave 32x64 (2x4 WMMA tiles),
// K staged 32-deep in double-buffered LDS. X staged via VGPR cvt (f32->bf16),
// W staged via GLOBAL_LOAD_ASYNC_TO_LDS_B128.
// ---------------------------------------------------------------------------
__global__ __launch_bounds__(256) void lora_gemm(
    const float* __restrict__ X,
    const unsigned short* __restrict__ Wq,
    const float* __restrict__ bias,
    float* __restrict__ Y)
{
    __shared__ __bf16 Xs[2][MTILE * LDSS];
    __shared__ __bf16 Ws[2][NTILE * LDSS];

    const __bf16* __restrict__ W = (const __bf16*)Wq;

    const int t    = threadIdx.x;
    const int lane = t & 31;
    const int wave = t >> 5;
    const int half = (lane >> 4) & 1;   // K-half selector per ISA lane layout
    const int l16  = lane & 15;

    const int blockM = blockIdx.y * MTILE;
    const int blockN = blockIdx.x * NTILE;

    const int waveM = (wave & 3) * 32;  // 4 waves along M
    const int waveN = (wave >> 2) * 64; // 2 waves along N

    // staging index: X tile 128x32 f32 (8 thr/row, 4 passes); W tile 128x32 bf16 (4 thr/row, 2 passes)
    const int xr = t >> 3;
    const int xc = (t & 7) * 4;
    const int wr = t >> 2;
    const int wc = (t & 3) * 8;

    v8f acc[2][4];
    #pragma unroll
    for (int mi = 0; mi < 2; ++mi)
        #pragma unroll
        for (int ni = 0; ni < 4; ++ni)
            #pragma unroll
            for (int e = 0; e < 8; ++e)
                acc[mi][ni][e] = 0.0f;

    float4 xreg[4];

    // ---- prologue: stage k-step 0 into buffer 0 ----
    #pragma unroll
    for (int p = 0; p < 4; ++p)
        xreg[p] = *(const float4*)(X + (size_t)(blockM + p * 32 + xr) * D_IN + xc);
    #pragma unroll
    for (int p = 0; p < 2; ++p)
        async_copy_b128((unsigned)(uintptr_t)(&Ws[0][(p * 64 + wr) * LDSS + wc]),
                        W + (size_t)(blockN + p * 64 + wr) * D_IN + wc);

    #pragma unroll
    for (int p = 0; p < 4; ++p) {
        v4bf pk;
        pk[0] = (__bf16)xreg[p].x; pk[1] = (__bf16)xreg[p].y;
        pk[2] = (__bf16)xreg[p].z; pk[3] = (__bf16)xreg[p].w;
        *(v4bf*)(&Xs[0][(p * 32 + xr) * LDSS + xc]) = pk;
    }
    wait_async0();
    __syncthreads();

    int buf = 0;
    for (int ks = 0; ks < KSTEPS; ++ks) {
        const bool more = (ks + 1) < KSTEPS;
        const int  nb   = buf ^ 1;

        // ---- issue next-stage loads early (latency hiding) ----
        if (more) {
            const int kn = (ks + 1) * KTILE;
            #pragma unroll
            for (int p = 0; p < 4; ++p)
                xreg[p] = *(const float4*)(X + (size_t)(blockM + p * 32 + xr) * D_IN + kn + xc);
            #pragma unroll
            for (int p = 0; p < 2; ++p)
                async_copy_b128((unsigned)(uintptr_t)(&Ws[nb][(p * 64 + wr) * LDSS + wc]),
                                W + (size_t)(blockN + p * 64 + wr) * D_IN + kn + wc);
        }

        // ---- compute on current buffer ----
        const __bf16* __restrict__ xsP = &Xs[buf][0];
        const __bf16* __restrict__ wsP = &Ws[buf][0];

        // A fragments: 16x32 bf16; lane l16 = row; halves hold K {0..7,16..23} / {8..15,24..31}
        v16bf afrag[2];
        #pragma unroll
        for (int mi = 0; mi < 2; ++mi) {
            const int arow = waveM + mi * 16 + l16;
            v8bf lo = *(const v8bf*)(xsP + arow * LDSS + half * 8);
            v8bf hi = *(const v8bf*)(xsP + arow * LDSS + 16 + half * 8);
            afrag[mi] = __builtin_shufflevector(lo, hi,
                0, 1, 2, 3, 4, 5, 6, 7, 8, 9, 10, 11, 12, 13, 14, 15);
        }

        // B fragments: 32x16 bf16; lane l16 = column (= W row); halves hold K {0..7,16..23} / {8..15,24..31}
        v16bf bfrag[4];
        #pragma unroll
        for (int ni = 0; ni < 4; ++ni) {
            const int brow = waveN + ni * 16 + l16;
            v8bf lo = *(const v8bf*)(wsP + brow * LDSS + half * 8);
            v8bf hi = *(const v8bf*)(wsP + brow * LDSS + 16 + half * 8);
            bfrag[ni] = __builtin_shufflevector(lo, hi,
                0, 1, 2, 3, 4, 5, 6, 7, 8, 9, 10, 11, 12, 13, 14, 15);
        }

        #pragma unroll
        for (int mi = 0; mi < 2; ++mi)
            #pragma unroll
            for (int ni = 0; ni < 4; ++ni)
                acc[mi][ni] = __builtin_amdgcn_wmma_f32_16x16x32_bf16(
                    /*neg_a=*/false, afrag[mi],
                    /*neg_b=*/false, bfrag[ni],
                    /*c_mod=*/(short)0, acc[mi][ni],
                    /*reuse_a=*/false, /*reuse_b=*/false);

        // ---- write next X stage into the other buffer; wait for async W ----
        if (more) {
            #pragma unroll
            for (int p = 0; p < 4; ++p) {
                v4bf pk;
                pk[0] = (__bf16)xreg[p].x; pk[1] = (__bf16)xreg[p].y;
                pk[2] = (__bf16)xreg[p].z; pk[3] = (__bf16)xreg[p].w;
                *(v4bf*)(&Xs[nb][(p * 32 + xr) * LDSS + xc]) = pk;
            }
            wait_async0();
        }

        __syncthreads();
        buf ^= 1;
    }

    // ---- epilogue: C layout (VGPR v, lanes 0-15: M=v, N=l16; lanes 16-31: M=v+8) ----
    #pragma unroll
    for (int mi = 0; mi < 2; ++mi) {
        const int gmBase = blockM + waveM + mi * 16 + half * 8;
        #pragma unroll
        for (int ni = 0; ni < 4; ++ni) {
            const int gn = blockN + waveN + ni * 16 + l16;
            const float bv = bias[gn];
            float* __restrict__ yp = Y + (size_t)gmBase * D_OUT + gn;
            #pragma unroll
            for (int v = 0; v < 8; ++v)
                yp[(size_t)v * D_OUT] = acc[mi][ni][v] + bv;
        }
    }
}

// ---------------------------------------------------------------------------
extern "C" void kernel_launch(void* const* d_in, const int* in_sizes, int n_in,
                              void* d_out, int out_size, void* d_ws, size_t ws_size,
                              hipStream_t stream) {
    const float* x       = (const float*)d_in[0];
    const float* W_base  = (const float*)d_in[1];
    const float* b_base  = (const float*)d_in[2];
    const float* A32     = (const float*)d_in[3];
    const float* B32     = (const float*)d_in[4];
    const float* A16     = (const float*)d_in[5];
    const float* B16     = (const float*)d_in[6];
    const float* A8      = (const float*)d_in[7];
    const float* B8      = (const float*)d_in[8];
    const float* A4      = (const float*)d_in[9];
    const float* B4      = (const float*)d_in[10];
    const float* A2      = (const float*)d_in[11];
    const float* B2      = (const float*)d_in[12];
    const float* A1      = (const float*)d_in[13];
    const float* B1      = (const float*)d_in[14];
    const float* lambdas = (const float*)d_in[15];
    const float* scaling = (const float*)d_in[16];

    unsigned short* Wq = (unsigned short*)d_ws;   // 1024*1024 bf16 = 2 MB
    float* y = (float*)d_out;

    fold_weights<<<(D_OUT * D_IN) / 256, 256, 0, stream>>>(
        W_base, A32, B32, A16, B16, A8, B8, A4, B4, A2, B2, A1, B1,
        lambdas, scaling, Wq);

    const int M = 4 * 8192;
    dim3 grid(D_OUT / NTILE, M / MTILE);   // 8 x 256 workgroups
    lora_gemm<<<grid, 256, 0, stream>>>(x, Wq, b_base, y);
}
